// MaskDecoderUniform_16844861735723
// MI455X (gfx1250) — matrix operand order repaired and checked by reading, not compiled, not run on hardware
//
#include <hip/hip_runtime.h>

// ---------------------------------------------------------------------------
// MaskDecoderUniform for MI455X (gfx1250, wave32, WMMA bf16 16x16x32)
//
// GroupNorm(1,C) per band -> fc1(128->512)+tanh -> fc2(512->12bw)+GLU
//   K=31 bands, B=4, C=128, T=2000, H=512.  ~58 GFLOP of GEMM on
//   v_wmma_f32_16x16x32_bf16 (bf16 operands, fp32 accumulate).
//
// Per-call prep:
//   - x transposed once (K innermost -> T innermost): all heavy loads coalesced
//   - fc1_w/fc2_w pre-packed to bf16 in WMMA A-fragment register order
//     ([k][mt][ks][lane][v] dwords) -> A fragment = 2x global_load_b128,
//     zero cvt/perm VALU in the GEMM inner loops
// tanh via gfx1250 v_tanh_f32; sigmoid via v_exp + v_rcp (branchless).
// bf16 pairs flow as bf16x2 end-to-end (no bit_cast) so the backend keeps a
// single v_cvt_pk_bf16_f32 per packed dword.
// Stats use fixed-order reductions (no float atomics) -> deterministic replay.
//
// Workspace, requires ws_size >= ~142 MB:
//   [0 .. 1024)                     stats: (mean, rstd) per (b,k)
//   [1024 .. +1015808)              partial (sum, sumsq)
//   [1016832 .. +31744000)          xT[k][b][c][t]  fp32
//   [32760832 .. +1015808)          w1p  (bf16-pair dwords)
//   [33776640 .. +1650688)          w2p  (bf16-pair dwords)
// ---------------------------------------------------------------------------

#define KB   31
#define CC   128
#define TT   2000
#define HH   512
#define MAXO 204          // 17*12
#define TTILE 64
#define NTT  32           // ceil(2000/64)
#define NBANDS_TOTAL 257
#define NMT2 13           // ceil(204/16): fc2 M-tiles (padded)

__constant__ int c_bw[KB]  = {2,3,3,3,3,3,3,3,3,3,3,8,8,8,8,8,8,8,8,8,8,8,8,
                              16,16,16,16,16,16,16,17};
__constant__ int c_off[KB] = {0,2,5,8,11,14,17,20,23,26,29,32,40,48,56,64,72,
                              80,88,96,104,112,120,128,144,160,176,192,208,224,240};

typedef __bf16 v16bf  __attribute__((ext_vector_type(16)));
typedef __bf16 bf16x2 __attribute__((ext_vector_type(2)));
typedef float  v8f    __attribute__((ext_vector_type(8)));

union Frag { unsigned u[8]; v16bf v; };

// One packed bf16 pair; keep it in vector form (no bit_cast) so isel emits a
// single v_cvt_pk_bf16_f32.
__device__ __forceinline__ bf16x2 cvt2(float a, float b) {
#if __has_builtin(__builtin_amdgcn_cvt_pk_bf16_f32)
    return __builtin_amdgcn_cvt_pk_bf16_f32(a, b);
#else
    bf16x2 r; r[0] = (__bf16)a; r[1] = (__bf16)b; return r;
#endif
}

__device__ __forceinline__ float tanh_fast(float x) {
#if __has_builtin(__builtin_amdgcn_tanhf)
    return __builtin_amdgcn_tanhf(x);                  // v_tanh_f32
#else
    float cx = fminf(fmaxf(x, -20.0f), 20.0f);
    float e  = __expf(2.0f * cx);
    return (e - 1.0f) * __builtin_amdgcn_rcpf(e + 1.0f);
#endif
}

__device__ __forceinline__ float sigmoid_fast(float x) {
    return __builtin_amdgcn_rcpf(1.0f + __expf(-x));
}

// A-fragment K-pair element offset within a 32-wide K step (contiguous order: k = 16*hi + 2v, as in the B reads):
__device__ __forceinline__ int a_frag_cb(int ks, int v, int hi) {
    return ks * 32 + (hi << 4) + (v << 1);
}

// ---------------------------------------------------------------------------
// Kernel 1: transpose x [B,C,T,K] -> xT [K,B,C,T] + deterministic partial
// (sum, sumsq) per (b,c,ttile,k). grid = B*C*NTT blocks of 256.
// ---------------------------------------------------------------------------
__global__ __launch_bounds__(256)
void k_transpose_stats(const float* __restrict__ x,
                       float* __restrict__ xT,
                       float* __restrict__ partials) {
    int bx  = blockIdx.x;
    int tc  = bx % NTT;
    int rem = bx / NTT;
    int c   = rem % CC;
    int b   = rem / CC;
    int t0  = tc * TTILE;
    int nvalid = TT - t0; if (nvalid > TTILE) nvalid = TTILE;

    __shared__ float tile[KB][TTILE + 1];
    __shared__ float s1[KB][8], s2[KB][8];

    const float* base = x + ((size_t)(b * CC + c) * TT + t0) * KB;
    int tid = threadIdx.x;

    for (int f = tid; f < KB * TTILE; f += 256) {       // contiguous reads
        int t = f / KB, k = f % KB;
        tile[k][t] = (t < nvalid) ? base[f] : 0.0f;
    }
    __syncthreads();

    for (int f = tid; f < KB * TTILE; f += 256) {       // contiguous writes
        int k = f >> 6, t = f & 63;
        if (t < nvalid)
            xT[(((size_t)k * 4 + b) * CC + c) * TT + t0 + t] = tile[k][t];
    }

    if (tid < KB * 8) {                                  // fixed-order partials
        int k = tid >> 3, sub = tid & 7;
        float a = 0.f, q = 0.f;
        for (int t = sub; t < nvalid; t += 8) { float v = tile[k][t]; a += v; q += v * v; }
        s1[k][sub] = a; s2[k][sub] = q;
    }
    __syncthreads();
    if (tid < KB) {
        float a = 0.f, q = 0.f;
        for (int j = 0; j < 8; ++j) { a += s1[tid][j]; q += s2[tid][j]; }
        size_t pi = (size_t)bx * KB + tid;
        partials[pi * 2] = a; partials[pi * 2 + 1] = q;
    }
}

// ---------------------------------------------------------------------------
// Kernel 2: reduce partials -> (mean, rstd) per (b,k). grid = 124, 256 thr.
// ---------------------------------------------------------------------------
__global__ __launch_bounds__(256)
void k_reduce_stats(const float* __restrict__ partials,
                    float* __restrict__ stats) {
    int g = blockIdx.x;                 // b*31 + k
    int b = g / KB, k = g % KB;
    __shared__ float r1[256], r2[256];
    int tid = threadIdx.x;
    float a = 0.f, q = 0.f;
    for (int pc = tid; pc < CC * NTT; pc += 256) {
        size_t pi = ((size_t)b * (CC * NTT) + pc) * KB + k;
        a += partials[pi * 2]; q += partials[pi * 2 + 1];
    }
    r1[tid] = a; r2[tid] = q;
    __syncthreads();
    for (int s = 128; s > 0; s >>= 1) {
        if (tid < s) { r1[tid] += r1[tid + s]; r2[tid] += r2[tid + s]; }
        __syncthreads();
    }
    if (tid == 0) {
        float inv  = 1.0f / (float)(CC * TT);
        float mean = r1[0] * inv;
        float var  = r2[0] * inv - mean * mean;
        stats[g * 2]     = mean;
        stats[g * 2 + 1] = rsqrtf(var + 1e-5f);
    }
}

// ---------------------------------------------------------------------------
// Kernel 3a/3b: pre-pack weights as bf16 pairs in A-fragment order:
//   w1p[k][mt(32)][ks(4)][lane(32)][v(8)]   (fc1: M=h, K=c)
//   w2p[k][mt(13)][ks(16)][lane(32)][v(8)]  (fc2: M=o, K=h; rows >= 12bw -> 0)
// ---------------------------------------------------------------------------
__global__ __launch_bounds__(256)
void k_pack_w1(const float* __restrict__ fc1_w, bf16x2* __restrict__ w1p) {
    int idx = blockIdx.x * 256 + threadIdx.x;          // 31*32*4*32*8
    if (idx >= KB * 32 * 4 * 32 * 8) return;
    int v    = idx & 7;
    int lane = (idx >> 3) & 31;
    int ks   = (idx >> 8) & 3;
    int mt   = (idx >> 10) & 31;
    int k    = idx >> 15;
    int M  = mt * 16 + (lane & 15);
    int cb = a_frag_cb(ks, v, lane >> 4);
    const float* s = fc1_w + ((size_t)k * HH + M) * CC + cb;
    w1p[idx] = cvt2(s[0], s[1]);
}

__global__ __launch_bounds__(256)
void k_pack_w2(const float* __restrict__ fc2_w, bf16x2* __restrict__ w2p) {
    int idx = blockIdx.x * 256 + threadIdx.x;          // 31*13*16*32*8
    if (idx >= KB * NMT2 * 16 * 32 * 8) return;
    int v    = idx & 7;
    int lane = (idx >> 3) & 31;
    int ks   = (idx >> 8) & 15;
    int km   = idx >> 12;
    int mt   = km % NMT2;
    int k    = km / NMT2;
    int M    = mt * 16 + (lane & 15);
    int O    = 12 * c_bw[k];
    bf16x2 val = cvt2(0.f, 0.f);
    if (M < O) {
        int hb2 = a_frag_cb(ks, v, lane >> 4);
        const float* s = fc2_w + ((size_t)k * MAXO + M) * HH + hb2;
        val = cvt2(s[0], s[1]);
    }
    w2p[idx] = val;
}

// ---------------------------------------------------------------------------
// Kernel 4: fused norm -> fc1(WMMA)+tanh -> fc2(WMMA) -> GLU -> out.
// One block per (k, b, 64-t tile). 256 threads = 8 waves.
// Dynamic LDS (120320 B):
//   [0)        xn pairs [64][65] bf16x2  -- reused as obuf [64][209] f32
//   [53504)    hbuf     [64][257] bf16x2 (h pairs)
//   [119296)   sc[128], sh[128]  f32
// ---------------------------------------------------------------------------
__global__ __launch_bounds__(256)
void k_main(const float* __restrict__ xT, const float* __restrict__ stats,
            const float* __restrict__ norm_w, const float* __restrict__ norm_b,
            const unsigned* __restrict__ w1p, const float* __restrict__ fc1_b,
            const unsigned* __restrict__ w2p, const float* __restrict__ fc2_b,
            float* __restrict__ out) {
    extern __shared__ char smem[];
    bf16x2*   xnv  = (bf16x2*)smem;                // [64][65]  (writer view)
    unsigned* xn   = (unsigned*)smem;              //           (reader view)
    float*    obuf = (float*)smem;                 // [64][209]
    bf16x2*   hbv  = (bf16x2*)(smem + 53504);      // [64][257] (writer view)
    unsigned* hbuf = (unsigned*)(smem + 53504);    //           (reader view)
    float*    sc   = (float*)(smem + 119296);      // [128]
    float*    sh   = sc + CC;                      // [128]

    int bx = blockIdx.x;
    int k  = bx / (4 * NTT);
    int r0 = bx % (4 * NTT);
    int b  = r0 / NTT;
    int tt = r0 % NTT;
    int t0 = tt * TTILE;

    int tid  = threadIdx.x;
    int lane = tid & 31;
    int wave = tid >> 5;
    int lo16 = lane & 15;
    int hi   = lane >> 4;

    float mean = stats[(b * KB + k) * 2];
    float rstd = stats[(b * KB + k) * 2 + 1];

    if (tid < CC) {
        float g  = norm_w[k * CC + tid];
        float be = norm_b[k * CC + tid];
        float s  = rstd * g;
        sc[tid] = s;
        sh[tid] = be - mean * s;
    }
    __syncthreads();

    // ---- stage normalized bf16 activations: xn[t][c/2] ----
    const float* xb = xT + ((size_t)(k * 4 + b) * CC) * TT;
    for (int i = tid; i < 64 * TTILE; i += 256) {
        int p = i >> 6;                 // channel pair
        int t = i & 63;
        int c0 = p * 2;
        int tg = t0 + t;
        float v0 = 0.f, v1 = 0.f;
        if (tg < TT) {
            v0 = xb[(size_t)c0 * TT + tg];
            v1 = xb[(size_t)(c0 + 1) * TT + tg];
        }
        v0 = v0 * sc[c0]     + sh[c0];
        v1 = v1 * sc[c0 + 1] + sh[c0 + 1];
        xnv[t * 65 + p] = cvt2(v0, v1);
    }
    __syncthreads();

    // ---- fc1: h[t][h] = tanh(xn @ W1^T + b1), 32 M-tiles over 8 waves ----
    for (int mi = 0; mi < 4; ++mi) {
        int mt = wave * 4 + mi;
        int h0 = mt * 16;
        Frag A[4];
        for (int ks = 0; ks < 4; ++ks) {
            const unsigned* wf = w1p + ((((size_t)k * 32 + mt) * 4 + ks) * 32 + lane) * 8;
            for (int v = 0; v < 8; ++v) A[ks].u[v] = wf[v];   // 2x b128
        }
        float bias[8];
        for (int r = 0; r < 8; ++r) bias[r] = fc1_b[k * HH + h0 + hi * 8 + r];

        for (int nt = 0; nt < 4; ++nt) {
            int tl = nt * 16 + lo16;
            v8f acc = {};
            for (int ks = 0; ks < 4; ++ks) {
                Frag B;
                int pb = ks * 16 + hi * 8;          // B: K pairs per lane half
                for (int v = 0; v < 8; ++v) B.u[v] = xn[tl * 65 + pb + v];
                acc = __builtin_amdgcn_wmma_f32_16x16x32_bf16(
                        false, A[ks].v, false, B.v, (short)0, acc, false, false);
            }
            bf16x2* hb = hbv + tl * 257 + ((h0 + hi * 8) >> 1);
            for (int j = 0; j < 4; ++j)
                hb[j] = cvt2(tanh_fast(acc[2 * j] + bias[2 * j]),
                             tanh_fast(acc[2 * j + 1] + bias[2 * j + 1]));
        }
    }
    __syncthreads();

    // ---- fc2: o[t][o] = h @ W2^T + b2 (pre-activation to obuf) ----
    int bw = c_bw[k];
    int O  = 12 * bw;
    int nmt = (O + 15) >> 4;
    for (int mt = wave; mt < nmt; mt += 8) {
        int o0 = mt * 16;
        v8f acc[4] = {};
        for (int ks = 0; ks < 16; ++ks) {
            Frag A;
            const unsigned* wf = w2p + ((((size_t)k * NMT2 + mt) * 16 + ks) * 32 + lane) * 8;
            for (int v = 0; v < 8; ++v) A.u[v] = wf[v];       // 2x b128
            for (int nt = 0; nt < 4; ++nt) {
                int tl = nt * 16 + lo16;
                Frag B;
                int pb = ks * 16 + hi * 8;
                for (int v = 0; v < 8; ++v) B.u[v] = hbuf[tl * 257 + pb + v];
                acc[nt] = __builtin_amdgcn_wmma_f32_16x16x32_bf16(
                        false, A.v, false, B.v, (short)0, acc[nt], false, false);
            }
        }
        for (int nt = 0; nt < 4; ++nt) {
            int tl = nt * 16 + lo16;
            for (int r = 0; r < 8; ++r) {
                int o = o0 + hi * 8 + r;
                if (o < O) obuf[tl * 209 + o] = acc[nt][r] + fc2_b[k * MAXO + o];
            }
        }
    }
    __syncthreads();

    // ---- GLU epilogue + coalesced store to out[b][257][T][3][2] ----
    int off  = c_off[k];
    int half = 6 * bw;
    int total = bw * TTILE * 6;
    for (int e = tid; e < total; e += 256) {
        int nl = e / (TTILE * 6);
        int r6 = e % (TTILE * 6);
        int t  = r6 / 6;
        int rr = r6 % 6;
        int tg = t0 + t;
        if (tg < TT) {
            float a = obuf[t * 209 + nl * 6 + rr];
            float g = obuf[t * 209 + half + nl * 6 + rr];
            out[((size_t)(b * NBANDS_TOTAL + off + nl) * TT + tg) * 6 + rr] =
                a * sigmoid_fast(g);
        }
    }
}

// ---------------------------------------------------------------------------
extern "C" void kernel_launch(void* const* d_in, const int* in_sizes, int n_in,
                              void* d_out, int out_size, void* d_ws, size_t ws_size,
                              hipStream_t stream) {
    const float* x      = (const float*)d_in[0];
    const float* norm_w = (const float*)d_in[1];
    const float* norm_b = (const float*)d_in[2];
    const float* fc1_w  = (const float*)d_in[3];
    const float* fc1_b  = (const float*)d_in[4];
    const float* fc2_w  = (const float*)d_in[5];
    const float* fc2_b  = (const float*)d_in[6];
    float* outp = (float*)d_out;

    const size_t N_PART = (size_t)4 * CC * NTT * KB * 2;       // 1,015,808
    const size_t N_XT   = (size_t)KB * 4 * CC * TT;            // 31,744,000
    const size_t N_W1P  = (size_t)KB * 32 * 4 * 32 * 8;        // 1,015,808
    const size_t N_W2P  = (size_t)KB * NMT2 * 16 * 32 * 8;     // 1,650,688

    float*  ws       = (float*)d_ws;
    float*  stats    = ws;
    float*  partials = ws + 1024;
    float*  xT       = ws + 1024 + N_PART;
    bf16x2* w1p      = (bf16x2*)(ws + 1024 + N_PART + N_XT);
    bf16x2* w2p      = w1p + N_W1P;

    (void)in_sizes; (void)n_in; (void)out_size; (void)ws_size;

    k_transpose_stats<<<4 * CC * NTT, 256, 0, stream>>>(x, xT, partials);
    k_reduce_stats<<<4 * KB, 256, 0, stream>>>(partials, stats);
    k_pack_w1<<<(int)((N_W1P + 255) / 256), 256, 0, stream>>>(fc1_w, w1p);
    k_pack_w2<<<(int)((N_W2P + 255) / 256), 256, 0, stream>>>(fc2_w, w2p);

    size_t lds = 53504 + 64 * 257 * 4 + 1024;   // 120320 B (< 320 KB/WGP)
    (void)hipFuncSetAttribute((const void*)k_main,
        hipFuncAttributeMaxDynamicSharedMemorySize, (int)lds);

    k_main<<<KB * 4 * NTT, 256, lds, stream>>>(xT, stats, norm_w, norm_b,
                                               (const unsigned*)w1p, fc1_b,
                                               (const unsigned*)w2p, fc2_b, outp);
}
